// FusionRQVAE_v2_47923245089213
// MI455X (gfx1250) — matrix operand order repaired and checked by reading, not compiled, not run on hardware
//
#include <hip/hip_runtime.h>
#include <hip/hip_bf16.h>
#include <cstdint>
#include <cstddef>

// ---------------------------------------------------------------------------
// Types for CDNA5 WMMA (gfx1250, wave32)
// ---------------------------------------------------------------------------
typedef __attribute__((ext_vector_type(16))) __bf16 bf16x16;
typedef __attribute__((ext_vector_type(8)))  float  v8f;
typedef __attribute__((ext_vector_type(4)))  int    i32x4;

union Frag {
    bf16x16 v;
    i32x4   q[2];
};

__device__ __forceinline__ unsigned short f2bf(float x) {
    unsigned int u = __float_as_uint(x);
    return (unsigned short)((u + 0x7fffu + ((u >> 16) & 1u)) >> 16);
}

// ---------------------------------------------------------------------------
// Generic bf16 WMMA GEMM:  C[M x N] = A[M x K](bf16) * W + bias (+Cadd) (ReLU?)
//  - A row-major bf16 (as ushort bit patterns)
//  - Wp: packed transposed weights, uint32 [N][K/2], dword = (W[2m][n] | W[2m+1][n]<<16)
//    This matches the B-fragment layout: lane n%16, 8 contiguous dwords = 16 K vals.
//  - block = 256 threads = 8 waves; each wave computes 16 rows x (NTILES*16) cols
//    via NTILES x v_wmma_f32_16x16x32_bf16 per K-chunk of 32.
//  - Requirements (guaranteed by host dispatch): M % 128 == 0, K % 32 == 0,
//    N % (NTILES*16) == 0.  -> fully branchless inner loop, no EXEC games.
// ---------------------------------------------------------------------------
template <int NTILES>
__global__ __launch_bounds__(256) void gemm_bf16_wmma(
    const unsigned short* __restrict__ A,
    const unsigned int*   __restrict__ Wp,
    const float* __restrict__ bias,   // may be null
    const float* __restrict__ Cadd,   // may be null, [M x N] f32
    float*          __restrict__ outF, // may be null
    unsigned short* __restrict__ outB, // may be null (bf16)
    int N, int K, int relu)
{
    const int wid  = threadIdx.x >> 5;   // 0..7
    const int lane = threadIdx.x & 31;
    const int l15  = lane & 15;
    const int lh   = lane >> 4;          // 0 or 1
    const int mbase = blockIdx.y * 128 + wid * 16;
    const int nbase = blockIdx.x * (NTILES * 16);
    const int Kh = K >> 1;

    v8f acc[NTILES] = {};

    float bv[NTILES];
#pragma unroll
    for (int nt = 0; nt < NTILES; ++nt) {
        int c = nbase + nt * 16 + l15;
        bv[nt] = bias ? bias[c] : 0.0f;
    }

    // Per-lane base pointers for the NTILES B-fragment streams (L2-resident).
    const unsigned int* bptr[NTILES];
#pragma unroll
    for (int nt = 0; nt < NTILES; ++nt) {
        bptr[nt] = Wp + (size_t)(nbase + nt * 16 + l15) * (size_t)Kh + lh * 8;
    }

    const unsigned short* Arow = A + (size_t)(mbase + l15) * (size_t)K;

    for (int kc = 0; kc < K; kc += 32) {
        Frag af;
        // A fragment as loaded here (contiguous k, same order as B): lanes 0-15 hold K {0..15},
        // lanes 16-31 hold K {16..31} of row M = lane&15.
        af.q[0] = *(const i32x4*)(Arow + kc + lh * 16);
        af.q[1] = *(const i32x4*)(Arow + kc + lh * 16 + 8);
        // Prefetch next A K-chunk (streams from HBM).
        __builtin_prefetch((const void*)(Arow + kc + 32), 0, 0);
#pragma unroll
        for (int nt = 0; nt < NTILES; ++nt) {
            Frag bfm;
            const unsigned int* bp = bptr[nt] + (kc >> 1);
            bfm.q[0] = *(const i32x4*)(bp);
            bfm.q[1] = *(const i32x4*)(bp + 4);
            acc[nt] = __builtin_amdgcn_wmma_f32_16x16x32_bf16(
                false, af.v, false, bfm.v, (short)0, acc[nt], false, false);
        }
    }

    // C/D layout: VGPR r -> row = mbase + lh*8 + r, col = nbase + nt*16 + l15
#pragma unroll
    for (int nt = 0; nt < NTILES; ++nt) {
        int col = nbase + nt * 16 + l15;
#pragma unroll
        for (int r = 0; r < 8; ++r) {
            int row = mbase + lh * 8 + r;
            float v = acc[nt][r] + bv[nt];
            size_t o = (size_t)row * (size_t)N + col;
            if (Cadd) v += Cadd[o];
            if (relu) v = fmaxf(v, 0.0f);
            if (outF) outF[o] = v;
            if (outB) outB[o] = f2bf(v);
        }
    }
}

// ---------------------------------------------------------------------------
// Pack weights: f32 W[Din x Dout] row-major -> uint32 Wp[Dout][Din/2]
// ---------------------------------------------------------------------------
__global__ void pack_w_kernel(const float* __restrict__ W,
                              unsigned int* __restrict__ Wp,
                              int Din, int Dout)
{
    int idx = blockIdx.x * blockDim.x + threadIdx.x;
    int Kh = Din >> 1;
    if (idx >= Dout * Kh) return;
    int n = idx / Kh;
    int m = idx - n * Kh;
    unsigned int lo = f2bf(W[(size_t)(2 * m)     * Dout + n]);
    unsigned int hi = f2bf(W[(size_t)(2 * m + 1) * Dout + n]);
    Wp[idx] = lo | (hi << 16);
}

// Fused LoRA matrix: M[k][j] = sum_r Bm[k][r] * Amat[r][j]  (64x32 @ 32x64),
// packed directly into the transposed bf16-pair layout Wp[j][k/2].
__global__ void lora_pack_kernel(const float* __restrict__ Bm,   // [64 x 32]
                                 const float* __restrict__ Amat, // [32 x 64]
                                 unsigned int* __restrict__ Wp)  // [64][32]
{
    int idx = blockIdx.x * blockDim.x + threadIdx.x;
    if (idx >= 64 * 32) return;
    int n = idx >> 5;   // output column j
    int m = idx & 31;   // K pair index
    float lo = 0.0f, hi = 0.0f;
    for (int r = 0; r < 32; ++r) {
        lo += Bm[(2 * m)     * 32 + r] * Amat[r * 64 + n];
        hi += Bm[(2 * m + 1) * 32 + r] * Amat[r * 64 + n];
    }
    Wp[idx] = (unsigned int)f2bf(lo) | ((unsigned int)f2bf(hi) << 16);
}

__global__ void f32_to_bf16_kernel(const float* __restrict__ in,
                                   unsigned short* __restrict__ out, int n)
{
    int i = blockIdx.x * blockDim.x + threadIdx.x;
    if (i < n) out[i] = f2bf(in[i]);
}

__global__ void zero_kernel(float* p, int n)
{
    int i = blockIdx.x * blockDim.x + threadIdx.x;
    if (i < n) p[i] = 0.0f;
}

// ---------------------------------------------------------------------------
// Residual VQ: one row (64 dims) per thread. Codebook [4][256][64] f32 staged
// in LDS 128 codes (32KB) at a time. Forward math:
//   q_st == q, residual_{l+1} = residual_l - q, x_q = z - residual_final,
//   loss_l = (1+beta)*mean((q - r_l)^2) = (1+beta)*mean(r_{l+1}^2)
// ---------------------------------------------------------------------------
__global__ __launch_bounds__(256) void rq_kernel(
    const float* __restrict__ z,     // [B x 64]
    const float* __restrict__ cb,    // [4*256*64]
    float*          __restrict__ xqF, // [B x 64]
    unsigned short* __restrict__ xqB, // [B x 64] bf16
    float* __restrict__ idx_out,      // [B x 4] (as float)
    float* __restrict__ loss_out)     // scalar (atomic)
{
    __shared__ float cbs[128 * 64];
    __shared__ float red[256];
    const int row = blockIdx.x * 256 + threadIdx.x;
    const float* zp = z + (size_t)row * 64;

    float r[64];
#pragma unroll
    for (int d = 0; d < 64; ++d) r[d] = zp[d];

    float lossAcc = 0.0f;
    int bidx[4];

    for (int l = 0; l < 4; ++l) {
        float best = 3.4e38f;
        int bi = 0;
        for (int half = 0; half < 2; ++half) {
            __syncthreads();
            const float* src = cb + ((size_t)l * 256 + half * 128) * 64;
            for (int t = threadIdx.x; t < 128 * 64; t += 256) cbs[t] = src[t];
            __syncthreads();
            for (int c = 0; c < 128; ++c) {
                const float* e = &cbs[c * 64];
                float dot = 0.0f, ee = 0.0f;
#pragma unroll
                for (int d = 0; d < 64; ++d) {
                    float ev = e[d];
                    dot = fmaf(r[d], ev, dot);
                    ee  = fmaf(ev, ev, ee);
                }
                // ||r||^2 constant per row -> argmin of (||e||^2 - 2 r.e)
                float dist = ee - 2.0f * dot;
                if (dist < best) { best = dist; bi = half * 128 + c; }
            }
        }
        bidx[l] = bi;
        const float* q = cb + ((size_t)l * 256 + bi) * 64;
        float pl = 0.0f;
#pragma unroll
        for (int d = 0; d < 64; ++d) {
            float nv = r[d] - q[d];
            r[d] = nv;
            pl = fmaf(nv, nv, pl);
        }
        lossAcc += pl;
    }

#pragma unroll
    for (int l = 0; l < 4; ++l) idx_out[(size_t)row * 4 + l] = (float)bidx[l];

    float* xo = xqF + (size_t)row * 64;
    unsigned short* xb = xqB + (size_t)row * 64;
#pragma unroll
    for (int d = 0; d < 64; ++d) {
        float v = zp[d] - r[d];  // x_q = z - final residual
        xo[d] = v;
        xb[d] = f2bf(v);
    }

    const float SCALE = (1.0f + 0.25f) / (4.0f * 65536.0f * 64.0f);
    red[threadIdx.x] = lossAcc * SCALE;
    __syncthreads();
    for (int s = 128; s > 0; s >>= 1) {
        if (threadIdx.x < s) red[threadIdx.x] += red[threadIdx.x + s];
        __syncthreads();
    }
    if (threadIdx.x == 0) atomicAdd(loss_out, red[0]);
}

// ---------------------------------------------------------------------------
// Host orchestration
// ---------------------------------------------------------------------------
extern "C" void kernel_launch(void* const* d_in, const int* in_sizes, int n_in,
                              void* d_out, int out_size, void* d_ws, size_t ws_size,
                              hipStream_t stream)
{
    (void)in_sizes; (void)n_in; (void)out_size; (void)ws_size;
    const int B = 65536;

    const float* xt = (const float*)d_in[0];
    const float* xv = (const float*)d_in[1];
    const float *teW[4], *teB[4], *tdW[4], *tdB[4];
    const float *veW[4], *veB[4], *vdW[4], *vdB[4];
    for (int i = 0; i < 4; ++i) {
        teW[i] = (const float*)d_in[2 + i];   teB[i] = (const float*)d_in[6 + i];
        tdW[i] = (const float*)d_in[10 + i];  tdB[i] = (const float*)d_in[14 + i];
        veW[i] = (const float*)d_in[18 + i];  veB[i] = (const float*)d_in[22 + i];
        vdW[i] = (const float*)d_in[26 + i];  vdB[i] = (const float*)d_in[30 + i];
    }
    const float* tcb = (const float*)d_in[34];
    const float* vcb = (const float*)d_in[35];
    const float* tA  = (const float*)d_in[36];
    const float* tBm = (const float*)d_in[37];
    const float* vA  = (const float*)d_in[38];
    const float* vBm = (const float*)d_in[39];

    const int encD[5] = {768, 1024, 512, 256, 64};
    const int decD[5] = {64, 256, 512, 1024, 768};

    // workspace bump allocator (256B aligned)
    char* base = (char*)d_ws;
    size_t off = 0;
    auto alloc = [&](size_t bytes) -> void* {
        void* p = base + off;
        off = (off + bytes + 255) & ~(size_t)255;
        return p;
    };

    unsigned int *wpTE[4], *wpTD[4], *wpVE[4], *wpVD[4];
    for (int i = 0; i < 4; ++i) wpTE[i] = (unsigned int*)alloc((size_t)encD[i + 1] * (encD[i] / 2) * 4);
    for (int i = 0; i < 4; ++i) wpTD[i] = (unsigned int*)alloc((size_t)decD[i + 1] * (decD[i] / 2) * 4);
    for (int i = 0; i < 4; ++i) wpVE[i] = (unsigned int*)alloc((size_t)encD[i + 1] * (encD[i] / 2) * 4);
    for (int i = 0; i < 4; ++i) wpVD[i] = (unsigned int*)alloc((size_t)decD[i + 1] * (decD[i] / 2) * 4);
    unsigned int* wpMvt = (unsigned int*)alloc(64 * 32 * 4);  // vis_Bm @ text_A
    unsigned int* wpMtv = (unsigned int*)alloc(64 * 32 * 4);  // text_Bm @ vis_A

    unsigned short* convB = (unsigned short*)alloc((size_t)B * 768 * 2);
    unsigned short* ping  = (unsigned short*)alloc((size_t)B * 1024 * 2);
    unsigned short* pong  = (unsigned short*)alloc((size_t)B * 512 * 2);
    float* zT  = (float*)alloc((size_t)B * 64 * 4);
    float* zV  = (float*)alloc((size_t)B * 64 * 4);
    float* xqFT = (float*)alloc((size_t)B * 64 * 4);
    float* xqFV = (float*)alloc((size_t)B * 64 * 4);
    unsigned short* xqBT = (unsigned short*)alloc((size_t)B * 64 * 2);
    unsigned short* xqBV = (unsigned short*)alloc((size_t)B * 64 * 2);
    unsigned short* decInT = (unsigned short*)alloc((size_t)B * 64 * 2);
    unsigned short* decInV = (unsigned short*)alloc((size_t)B * 64 * 2);

    // output layout (all f32): out_text, out_vis, loss_t, loss_v, idx_t, idx_v
    float* outF32 = (float*)d_out;
    float* outT  = outF32;
    float* outV  = outF32 + (size_t)B * 768;
    float* lossT = outV + (size_t)B * 768;
    float* lossV = lossT + 1;
    float* idxT  = lossV + 1;
    float* idxV  = idxT + (size_t)B * 4;

    zero_kernel<<<1, 64, 0, stream>>>(lossT, 2);

    auto pack = [&](const float* W, unsigned int* Wp, int Din, int Dout) {
        int total = Dout * (Din / 2);
        pack_w_kernel<<<(total + 255) / 256, 256, 0, stream>>>(W, Wp, Din, Dout);
    };
    for (int i = 0; i < 4; ++i) {
        pack(teW[i], wpTE[i], encD[i], encD[i + 1]);
        pack(tdW[i], wpTD[i], decD[i], decD[i + 1]);
        pack(veW[i], wpVE[i], encD[i], encD[i + 1]);
        pack(vdW[i], wpVD[i], decD[i], decD[i + 1]);
    }
    lora_pack_kernel<<<8, 256, 0, stream>>>(vBm, tA, wpMvt);
    lora_pack_kernel<<<8, 256, 0, stream>>>(tBm, vA, wpMtv);

    // Dispatch: N % 128 == 0 -> 8 column tiles per block; N == 64 -> 4 tiles.
    auto gemm = [&](const unsigned short* A, const unsigned int* Wp, const float* bias,
                    const float* Cadd, float* oF, unsigned short* oB,
                    int N, int K, int relu) {
        if ((N & 127) == 0) {
            dim3 grid(N / 128, B / 128);
            gemm_bf16_wmma<8><<<grid, 256, 0, stream>>>(A, Wp, bias, Cadd, oF, oB, N, K, relu);
        } else {
            dim3 grid(N / 64, B / 128);
            gemm_bf16_wmma<4><<<grid, 256, 0, stream>>>(A, Wp, bias, Cadd, oF, oB, N, K, relu);
        }
    };

    const int nconv = B * 768;

    // text encoder
    f32_to_bf16_kernel<<<(nconv + 255) / 256, 256, 0, stream>>>(xt, convB, nconv);
    gemm(convB, wpTE[0], teB[0], nullptr, nullptr, ping, 1024, 768, 1);
    gemm(ping,  wpTE[1], teB[1], nullptr, nullptr, pong, 512, 1024, 1);
    gemm(pong,  wpTE[2], teB[2], nullptr, nullptr, ping, 256, 512, 1);
    gemm(ping,  wpTE[3], teB[3], nullptr, zT, nullptr, 64, 256, 0);

    // vis encoder
    f32_to_bf16_kernel<<<(nconv + 255) / 256, 256, 0, stream>>>(xv, convB, nconv);
    gemm(convB, wpVE[0], veB[0], nullptr, nullptr, ping, 1024, 768, 1);
    gemm(ping,  wpVE[1], veB[1], nullptr, nullptr, pong, 512, 1024, 1);
    gemm(pong,  wpVE[2], veB[2], nullptr, nullptr, ping, 256, 512, 1);
    gemm(ping,  wpVE[3], veB[3], nullptr, zV, nullptr, 64, 256, 0);

    // residual VQ
    rq_kernel<<<B / 256, 256, 0, stream>>>(zT, tcb, xqFT, xqBT, idxT, lossT);
    rq_kernel<<<B / 256, 256, 0, stream>>>(zV, vcb, xqFV, xqBV, idxV, lossV);

    // cross-modal LoRA fusion: fused_t = xq_t + xq_v @ (vis_Bm @ text_A), etc.
    gemm(xqBV, wpMvt, nullptr, xqFT, nullptr, decInT, 64, 64, 0);
    gemm(xqBT, wpMtv, nullptr, xqFV, nullptr, decInV, 64, 64, 0);

    // text decoder
    gemm(decInT, wpTD[0], tdB[0], nullptr, nullptr, ping, 256, 64, 1);
    gemm(ping,   wpTD[1], tdB[1], nullptr, nullptr, pong, 512, 256, 1);
    gemm(pong,   wpTD[2], tdB[2], nullptr, nullptr, ping, 1024, 512, 1);
    gemm(ping,   wpTD[3], tdB[3], nullptr, outT, nullptr, 768, 1024, 0);

    // vis decoder
    gemm(decInV, wpVD[0], vdB[0], nullptr, nullptr, ping, 256, 64, 1);
    gemm(ping,   wpVD[1], vdB[1], nullptr, nullptr, pong, 512, 256, 1);
    gemm(pong,   wpVD[2], vdB[2], nullptr, nullptr, ping, 1024, 512, 1);
    gemm(ping,   wpVD[3], vdB[3], nullptr, outV, nullptr, 768, 1024, 0);
}
